// DecoderBlock_35081292874277
// MI455X (gfx1250) — matrix operand order repaired and checked by reading, not compiled, not run on hardware
//
#include <hip/hip_runtime.h>
#include <hip/hip_bf16.h>

// ---------- types ----------
typedef __attribute__((ext_vector_type(16))) __bf16 v16bf;
typedef __attribute__((ext_vector_type(8)))  __bf16 v8bf;
typedef __attribute__((ext_vector_type(8)))  float  v8f;

union frag16 { v16bf v; v8bf h[2]; };

__device__ inline unsigned short f32_bf16(float f) {
    unsigned int u = __float_as_uint(f);
    u = (u + 0x7FFFu + ((u >> 16) & 1u)) >> 16;
    return (unsigned short)u;
}

__device__ inline v8f vzero8() {
    v8f z;
#pragma unroll
    for (int i = 0; i < 8; ++i) z[i] = 0.f;
    return z;
}

// low 32 bits of a flat shared address == LDS byte offset (aperture rule)
__device__ inline unsigned int lds_off(const void* p) {
    return (unsigned int)(unsigned long long)p;
}

// CDNA5 async global->LDS copy, 16 bytes per lane, tracked by ASYNCcnt
__device__ inline void async_ld_b128(unsigned int ldsaddr, const void* gptr) {
    asm volatile("global_load_async_to_lds_b128 %0, %1, off"
                 :: "v"(ldsaddr), "v"(gptr)
                 : "memory");
}
__device__ inline void wait_async0() {
    asm volatile("s_wait_asynccnt 0" ::: "memory");
}

// ---------- elementwise conversion ----------
__global__ __launch_bounds__(256)
void cvt_kernel(const float* __restrict__ in, unsigned short* __restrict__ out, int n) {
    int i = blockIdx.x * 256 + threadIdx.x;
    if (i < n) out[i] = f32_bf16(in[i]);
}

// f32 [Rm x Cn] (batched) -> bf16 transposed [Cn x Rm]; 32x32 LDS tiles
__global__ __launch_bounds__(256)
void cvtT_kernel(const float* __restrict__ in, unsigned short* __restrict__ out,
                 int Rm, int Cn, long long inBS, long long outBS)
{
    __shared__ float tile[32][33];
    const float* inp = in + (size_t)blockIdx.z * inBS;
    unsigned short* outp = out + (size_t)blockIdx.z * outBS;
    const int bx = blockIdx.x * 32;   // Cn base
    const int by = blockIdx.y * 32;   // Rm base
    const int tx = threadIdx.x & 31;
    const int ty = threadIdx.x >> 5;  // 0..7
#pragma unroll
    for (int j = 0; j < 32; j += 8)
        tile[ty + j][tx] = inp[(size_t)(by + ty + j) * Cn + bx + tx];
    __syncthreads();
#pragma unroll
    for (int j = 0; j < 32; j += 8)
        outp[(size_t)(bx + ty + j) * Rm + by + tx] = f32_bf16(tile[tx][ty + j]);
}

// ---------- GEMM: C[R x Cc] = bf16 A[R x Kd] * (BT[Cc x Kd])^T + bias ----------
// 128x128 workgroup tile, 8 waves of 32x64, K-chunks of 32, async double-buffered LDS.
#define GPAD 40
__global__ __launch_bounds__(256)
void gemm_kernel(const unsigned short* __restrict__ A,
                 const unsigned short* __restrict__ BT,
                 const float* __restrict__ bias,
                 float* __restrict__ outF,
                 unsigned short* __restrict__ outB,
                 unsigned short* __restrict__ outBT,
                 int R, int Kd, int Cc, int relu)
{
    __shared__ unsigned short As[2][128][GPAD];   // [row][k]
    __shared__ unsigned short Bs[2][128][GPAD];   // [col][k]

    const int tid  = threadIdx.x;
    const int lane = tid & 31;
    const int wv   = tid >> 5;
    const int wr   = wv >> 1;      // 0..3 -> 32-row band
    const int wc   = wv & 1;       // 0..1 -> 64-col band
    const int rt0  = blockIdx.y * 128;
    const int ct0  = blockIdx.x * 128;

    const int mrow    = lane & 15;
    const int hi      = lane >> 4;
    const int a_base0 = hi ? 8 : 0;
    const int b_k0    = hi ? 8 : 0;

    v8f acc[2][4];
#pragma unroll
    for (int a = 0; a < 2; ++a)
#pragma unroll
        for (int b = 0; b < 4; ++b) acc[a][b] = vzero8();

    auto stage = [&](int k0, int buf) {
#pragma unroll
        for (int j = 0; j < 2; ++j) {
            const int idx = tid + j * 256;        // 0..511
            const int r  = idx >> 2;              // 0..127
            const int c8 = (idx & 3) * 8;         // 0,8,16,24
            async_ld_b128(lds_off(&As[buf][r][c8]),
                          A + (size_t)(rt0 + r) * Kd + k0 + c8);
            async_ld_b128(lds_off(&Bs[buf][r][c8]),
                          BT + (size_t)(ct0 + r) * Kd + k0 + c8);
        }
    };

    const int nchunk = Kd >> 5;
    stage(0, 0);
    for (int kc = 0; kc < nchunk; ++kc) {
        wait_async0();
        __syncthreads();
        if (kc + 1 < nchunk) stage((kc + 1) << 5, (kc + 1) & 1);
        const int buf = kc & 1;

        frag16 af[2], bfr[4];
#pragma unroll
        for (int ar = 0; ar < 2; ++ar) {
            const int rr = wr * 32 + ar * 16 + mrow;
            af[ar].h[0] = *(const v8bf*)&As[buf][rr][a_base0];
            af[ar].h[1] = *(const v8bf*)&As[buf][rr][a_base0 + 16];
        }
#pragma unroll
        for (int bc = 0; bc < 4; ++bc) {
            const int cc = wc * 64 + bc * 16 + mrow;
            bfr[bc].h[0] = *(const v8bf*)&Bs[buf][cc][b_k0];
            bfr[bc].h[1] = *(const v8bf*)&Bs[buf][cc][b_k0 + 16];
        }
#pragma unroll
        for (int ar = 0; ar < 2; ++ar)
#pragma unroll
            for (int bc = 0; bc < 4; ++bc)
                acc[ar][bc] = __builtin_amdgcn_wmma_f32_16x16x32_bf16(
                    false, af[ar].v, false, bfr[bc].v, (short)0, acc[ar][bc],
                    false, false);
    }

#pragma unroll
    for (int ar = 0; ar < 2; ++ar) {
#pragma unroll
        for (int bc = 0; bc < 4; ++bc) {
            const int col = ct0 + wc * 64 + bc * 16 + mrow;
            const float bb = bias ? bias[col] : 0.f;
#pragma unroll
            for (int i = 0; i < 8; ++i) {
                const int row = rt0 + wr * 32 + ar * 16 + i + 8 * hi;
                float v = acc[ar][bc][i] + bb;
                if (relu) v = fmaxf(v, 0.f);
                if (outF)  outF[(size_t)row * Cc + col] = v;
                if (outB)  outB[(size_t)row * Cc + col] = f32_bf16(v);
                if (outBT) outBT[(size_t)col * R + row] = f32_bf16(v);
            }
        }
    }
}

// ---------- flash attention ----------
// one block per (qblock=64 rows, head, batch); 4 waves x 16 rows.
// K staged natural [key][d], V staged from pre-transposed VT [n? see layout] via async.
#define APAD 72
__global__ __launch_bounds__(128)
void attn_kernel(const unsigned short* __restrict__ Qb,   // [tok][1024] bf16
                 const unsigned short* __restrict__ Kb,   // [tok][1024] bf16
                 const unsigned short* __restrict__ VT,   // [1024 cols][4096 tok] bf16
                 unsigned short* __restrict__ Yb,         // [tok][1024] bf16
                 int causal)
{
    __shared__ unsigned short Ks[2][64][APAD];    // [key][d]
    __shared__ unsigned short Vs[2][64][APAD];    // [d][key]
    __shared__ unsigned short Ps[4][16][APAD];    // per-wave P [row][key]

    const int tid  = threadIdx.x;
    const int lane = tid & 31;
    const int w    = tid >> 5;
    const int qblk = blockIdx.x;   // 0..15
    const int h    = blockIdx.y;   // 0..15
    const int n    = blockIdx.z;   // 0..3

    const int SEQ = 1024, MDL = 1024, TOK = 4096;
    const int D0 = h * 64;
    const size_t bh = (size_t)n * SEQ * MDL;
    const int tok0 = n * SEQ;

    const int mrow    = lane & 15;
    const int hi      = lane >> 4;
    const int a_base0 = hi ? 8 : 0;
    const int b_k0    = hi ? 8 : 0;

    // Q fragments: 16 rows x 64 d, two K=32 chunks; loaded once from global
    frag16 qa[2];
    const int qrow = qblk * 64 + w * 16 + mrow;
    const unsigned short* qptr = Qb + bh + (size_t)qrow * MDL + D0;
#pragma unroll
    for (int c = 0; c < 2; ++c) {
        qa[c].h[0] = *(const v8bf*)(qptr + c * 32 + a_base0);
        qa[c].h[1] = *(const v8bf*)(qptr + c * 32 + a_base0 + 16);
    }

    float mstat[8], lstat[8];
    v8f o[4];
#pragma unroll
    for (int i = 0; i < 8; ++i) { mstat[i] = -3.0e38f; lstat[i] = 0.f; }
#pragma unroll
    for (int dt = 0; dt < 4; ++dt) o[dt] = vzero8();

    auto stage = [&](int kt, int buf) {
        const int kb = kt * 64;
#pragma unroll
        for (int j = 0; j < 4; ++j) {
            const int idx = tid + j * 128;        // 0..511
            const int a  = idx >> 3;              // 0..63
            const int b8 = (idx & 7) * 8;         // 0..56
            async_ld_b128(lds_off(&Ks[buf][a][b8]),
                          Kb + bh + (size_t)(kb + a) * MDL + D0 + b8);
            async_ld_b128(lds_off(&Vs[buf][a][b8]),
                          VT + (size_t)(D0 + a) * TOK + tok0 + kb + b8);
        }
    };

    const int qr0 = qblk * 64 + w * 16;
    const int ntiles = causal ? (qblk + 1) : 16;

    stage(0, 0);
    for (int kt = 0; kt < ntiles; ++kt) {
        wait_async0();
        __syncthreads();
        if (kt + 1 < ntiles) stage(kt + 1, (kt + 1) & 1);
        const int buf = kt & 1;
        const int kb = kt * 64;

        // S = Q * K^T (16 rows x 64 keys per wave)
        v8f s[4];
#pragma unroll
        for (int nt = 0; nt < 4; ++nt) s[nt] = vzero8();
#pragma unroll
        for (int c = 0; c < 2; ++c) {
#pragma unroll
            for (int nt = 0; nt < 4; ++nt) {
                frag16 kf;
                const int key = nt * 16 + mrow;
                kf.h[0] = *(const v8bf*)&Ks[buf][key][c * 32 + b_k0];
                kf.h[1] = *(const v8bf*)&Ks[buf][key][c * 32 + b_k0 + 16];
                s[nt] = __builtin_amdgcn_wmma_f32_16x16x32_bf16(
                    false, qa[c].v, false, kf.v, (short)0, s[nt], false, false);
            }
        }

        // scale + causal mask + per-row tile max
        float tmax[8];
#pragma unroll
        for (int i = 0; i < 8; ++i) tmax[i] = -3.0e38f;
#pragma unroll
        for (int nt = 0; nt < 4; ++nt) {
            const int kc = kb + nt * 16 + mrow;
#pragma unroll
            for (int i = 0; i < 8; ++i) {
                float v = s[nt][i] * 0.125f;   // 1/sqrt(64)
                const int row = qr0 + i + 8 * hi;
                if (causal && kc > row) v = -3.0e38f;
                s[nt][i] = v;
                tmax[i] = fmaxf(tmax[i], v);
            }
        }
#pragma unroll
        for (int i = 0; i < 8; ++i) {
            float t = tmax[i];
            t = fmaxf(t, __shfl_xor(t, 1, 32));
            t = fmaxf(t, __shfl_xor(t, 2, 32));
            t = fmaxf(t, __shfl_xor(t, 4, 32));
            t = fmaxf(t, __shfl_xor(t, 8, 32));
            tmax[i] = t;
        }

        // online softmax update
        float tsum[8];
#pragma unroll
        for (int i = 0; i < 8; ++i) {
            const float mnew = fmaxf(mstat[i], tmax[i]);
            const float sc = __expf(mstat[i] - mnew);
            lstat[i] *= sc;
#pragma unroll
            for (int dt = 0; dt < 4; ++dt) o[dt][i] *= sc;
            mstat[i] = mnew;
            tsum[i] = 0.f;
        }
#pragma unroll
        for (int nt = 0; nt < 4; ++nt) {
#pragma unroll
            for (int i = 0; i < 8; ++i) {
                const float p = __expf(s[nt][i] - mstat[i]);
                tsum[i] += p;
                Ps[w][i + 8 * hi][nt * 16 + mrow] = f32_bf16(p);
            }
        }
#pragma unroll
        for (int i = 0; i < 8; ++i) {
            float t = tsum[i];
            t += __shfl_xor(t, 1, 32);
            t += __shfl_xor(t, 2, 32);
            t += __shfl_xor(t, 4, 32);
            t += __shfl_xor(t, 8, 32);
            lstat[i] += t;
        }

        // O += P * V
#pragma unroll
        for (int c = 0; c < 2; ++c) {
            frag16 pf;
            pf.h[0] = *(const v8bf*)&Ps[w][mrow][c * 32 + a_base0];
            pf.h[1] = *(const v8bf*)&Ps[w][mrow][c * 32 + a_base0 + 16];
#pragma unroll
            for (int dt = 0; dt < 4; ++dt) {
                frag16 vf;
                const int dcol = dt * 16 + mrow;
                vf.h[0] = *(const v8bf*)&Vs[buf][dcol][c * 32 + b_k0];
                vf.h[1] = *(const v8bf*)&Vs[buf][dcol][c * 32 + b_k0 + 16];
                o[dt] = __builtin_amdgcn_wmma_f32_16x16x32_bf16(
                    false, pf.v, false, vf.v, (short)0, o[dt], false, false);
            }
        }
    }

    // normalize + write concat-head layout y[tok][h*64+d] (bf16)
#pragma unroll
    for (int dt = 0; dt < 4; ++dt) {
#pragma unroll
        for (int i = 0; i < 8; ++i) {
            const float v = o[dt][i] / lstat[i];
            const int row = qr0 + i + 8 * hi;
            Yb[bh + (size_t)row * MDL + D0 + dt * 16 + mrow] = f32_bf16(v);
        }
    }
}

// ---------- residual + layernorm (row = 1024) ----------
__global__ __launch_bounds__(256)
void ln_kernel(const float* __restrict__ a, const float* __restrict__ b,
               const float* __restrict__ g, const float* __restrict__ be,
               float* __restrict__ outF, unsigned short* __restrict__ outB)
{
    const int Cc = 1024;
    const int row = blockIdx.x;
    const int tid = threadIdx.x;
    __shared__ float red[2][8];
    float x[4];
    float s = 0.f, s2 = 0.f;
#pragma unroll
    for (int j = 0; j < 4; ++j) {
        const int c = tid + j * 256;
        const float v = a[(size_t)row * Cc + c] + b[(size_t)row * Cc + c];
        x[j] = v; s += v; s2 += v * v;
    }
#pragma unroll
    for (int off = 16; off > 0; off >>= 1) {
        s  += __shfl_xor(s,  off, 32);
        s2 += __shfl_xor(s2, off, 32);
    }
    if ((tid & 31) == 0) { red[0][tid >> 5] = s; red[1][tid >> 5] = s2; }
    __syncthreads();
    float ts = 0.f, ts2 = 0.f;
#pragma unroll
    for (int i = 0; i < 8; ++i) { ts += red[0][i]; ts2 += red[1][i]; }
    const float mean = ts * (1.f / 1024.f);
    const float var  = ts2 * (1.f / 1024.f) - mean * mean;
    const float r    = rsqrtf(var + 1e-5f);
#pragma unroll
    for (int j = 0; j < 4; ++j) {
        const int c = tid + j * 256;
        const float v = (x[j] - mean) * r * g[c] + be[c];
        outF[(size_t)row * Cc + c] = v;
        if (outB) outB[(size_t)row * Cc + c] = f32_bf16(v);
    }
}

// ---------- host-side orchestration ----------
extern "C" void kernel_launch(void* const* d_in, const int* in_sizes, int n_in,
                              void* d_out, int out_size, void* d_ws, size_t ws_size,
                              hipStream_t stream)
{
    (void)in_sizes; (void)n_in; (void)out_size; (void)ws_size;
    const float* dec  = (const float*)d_in[0];
    const float* enc  = (const float*)d_in[1];
    // d_in[2] = mask: lower-triangular causal, handled analytically
    const float* Wq_s = (const float*)d_in[3];  const float* bq_s = (const float*)d_in[4];
    const float* Wk_s = (const float*)d_in[5];  const float* bk_s = (const float*)d_in[6];
    const float* Wv_s = (const float*)d_in[7];  const float* bv_s = (const float*)d_in[8];
    const float* Wo_s = (const float*)d_in[9];  const float* bo_s = (const float*)d_in[10];
    const float* Wq_c = (const float*)d_in[11]; const float* bq_c = (const float*)d_in[12];
    const float* Wk_c = (const float*)d_in[13]; const float* bk_c = (const float*)d_in[14];
    const float* Wv_c = (const float*)d_in[15]; const float* bv_c = (const float*)d_in[16];
    const float* Wo_c = (const float*)d_in[17]; const float* bo_c = (const float*)d_in[18];
    const float* W1 = (const float*)d_in[19];   const float* b1 = (const float*)d_in[20];
    const float* W2 = (const float*)d_in[21];   const float* b2 = (const float*)d_in[22];
    const float* g1 = (const float*)d_in[23];   const float* be1 = (const float*)d_in[24];
    const float* g2 = (const float*)d_in[25];   const float* be2 = (const float*)d_in[26];
    const float* g3 = (const float*)d_in[27];   const float* be3 = (const float*)d_in[28];

    char* ws = (char*)d_ws;
    size_t off = 0;
    const size_t MB = 1u << 20;
    auto carve = [&](size_t bytes) { void* p = ws + off; off += bytes; return p; };

    unsigned short* xb   = (unsigned short*)carve(8 * MB);   // dec bf16
    unsigned short* encb = (unsigned short*)carve(8 * MB);   // enc bf16
    unsigned short* wbuf = (unsigned short*)carve(8 * MB);   // weight staging (BT layouts)
    unsigned short* qb   = (unsigned short*)carve(8 * MB);
    unsigned short* kb   = (unsigned short*)carve(8 * MB);
    unsigned short* vbT  = (unsigned short*)carve(8 * MB);   // V transposed [1024][4096]
    unsigned short* yb   = (unsigned short*)carve(8 * MB);
    unsigned short* actb = (unsigned short*)carve(8 * MB);   // out2/out4 bf16
    float* tmpf = (float*)carve(16 * MB);                    // projection outs
    float* r1   = (float*)carve(16 * MB);                    // out2 f32
    float* r2   = (float*)carve(16 * MB);                    // out4 f32
    unsigned short* hid = (unsigned short*)carve(32 * MB);   // FFN hidden bf16

    const int TOK = 4096;               // N*K
    const int NE  = TOK * 1024;
    const size_t WOFF = 1u << 20;       // 1M bf16 elements = 2MB

    dim3 blk256(256), blk128(128);
    dim3 gProj(1024 / 128, TOK / 128);      // C=1024: 8 x 32
    dim3 gFfn1(4096 / 128, TOK / 128);      // C=4096: 32 x 32
    dim3 gAttn(16, 16, 4);
    dim3 gWqkv(64 / 32, 1024 / 32, 16);     // per-head [1024x64] transpose
    dim3 gW1k(1024 / 32, 1024 / 32, 1);
    dim3 gWf1(4096 / 32, 1024 / 32, 1);
    dim3 gWf2(1024 / 32, 4096 / 32, 1);

    // activations -> bf16
    cvt_kernel<<<NE / 256, blk256, 0, stream>>>(dec, xb, NE);
    cvt_kernel<<<NE / 256, blk256, 0, stream>>>(enc, encb, NE);

    // ---- self attention ----
    cvtT_kernel<<<gWqkv, blk256, 0, stream>>>(Wq_s, wbuf,            1024, 64, 65536, 65536);
    cvtT_kernel<<<gWqkv, blk256, 0, stream>>>(Wk_s, wbuf + WOFF,     1024, 64, 65536, 65536);
    cvtT_kernel<<<gWqkv, blk256, 0, stream>>>(Wv_s, wbuf + 2 * WOFF, 1024, 64, 65536, 65536);
    gemm_kernel<<<gProj, blk256, 0, stream>>>(xb, wbuf,            bq_s, nullptr, qb, nullptr, TOK, 1024, 1024, 0);
    gemm_kernel<<<gProj, blk256, 0, stream>>>(xb, wbuf + WOFF,     bk_s, nullptr, kb, nullptr, TOK, 1024, 1024, 0);
    gemm_kernel<<<gProj, blk256, 0, stream>>>(xb, wbuf + 2 * WOFF, bv_s, nullptr, nullptr, vbT, TOK, 1024, 1024, 0);
    attn_kernel<<<gAttn, blk128, 0, stream>>>(qb, kb, vbT, yb, 1);
    cvtT_kernel<<<gW1k, blk256, 0, stream>>>(Wo_s, wbuf, 1024, 1024, 0, 0);
    gemm_kernel<<<gProj, blk256, 0, stream>>>(yb, wbuf, bo_s, tmpf, nullptr, nullptr, TOK, 1024, 1024, 0);
    ln_kernel<<<TOK, blk256, 0, stream>>>(tmpf, dec, g1, be1, r1, actb);

    // ---- cross attention ----
    cvtT_kernel<<<gWqkv, blk256, 0, stream>>>(Wq_c, wbuf,            1024, 64, 65536, 65536);
    cvtT_kernel<<<gWqkv, blk256, 0, stream>>>(Wk_c, wbuf + WOFF,     1024, 64, 65536, 65536);
    cvtT_kernel<<<gWqkv, blk256, 0, stream>>>(Wv_c, wbuf + 2 * WOFF, 1024, 64, 65536, 65536);
    gemm_kernel<<<gProj, blk256, 0, stream>>>(actb, wbuf,            bq_c, nullptr, qb, nullptr, TOK, 1024, 1024, 0);
    gemm_kernel<<<gProj, blk256, 0, stream>>>(encb, wbuf + WOFF,     bk_c, nullptr, kb, nullptr, TOK, 1024, 1024, 0);
    gemm_kernel<<<gProj, blk256, 0, stream>>>(encb, wbuf + 2 * WOFF, bv_c, nullptr, nullptr, vbT, TOK, 1024, 1024, 0);
    attn_kernel<<<gAttn, blk128, 0, stream>>>(qb, kb, vbT, yb, 0);
    cvtT_kernel<<<gW1k, blk256, 0, stream>>>(Wo_c, wbuf, 1024, 1024, 0, 0);
    gemm_kernel<<<gProj, blk256, 0, stream>>>(yb, wbuf, bo_c, tmpf, nullptr, nullptr, TOK, 1024, 1024, 0);
    ln_kernel<<<TOK, blk256, 0, stream>>>(tmpf, r1, g2, be2, r2, actb);

    // ---- feedforward ----
    cvtT_kernel<<<gWf1, blk256, 0, stream>>>(W1, wbuf, 1024, 4096, 0, 0);  // -> W1T [4096x1024]
    gemm_kernel<<<gFfn1, blk256, 0, stream>>>(actb, wbuf, b1, nullptr, hid, nullptr, TOK, 1024, 4096, 1);
    cvtT_kernel<<<gWf2, blk256, 0, stream>>>(W2, wbuf, 4096, 1024, 0, 0);  // -> W2T [1024x4096]
    gemm_kernel<<<gProj, blk256, 0, stream>>>(hid, wbuf, b2, tmpf, nullptr, nullptr, TOK, 4096, 1024, 0);
    ln_kernel<<<TOK, blk256, 0, stream>>>(tmpf, r2, g3, be3, (float*)d_out, nullptr);
}
